// Fp8Linear_7069516169230
// MI455X (gfx1250) — matrix operand order repaired and checked by reading, not compiled, not run on hardware
//
#include <hip/hip_runtime.h>

// ---------------------------------------------------------------------------
// Fp8Linear: y = x @ bf16(W_fp8)^T + bias     (M=8192, N=3072, K=3072)
// Path A: (1) upcast W fp8->bf16 into d_ws, (2) pure-bf16 WMMA GEMM,
//         128x256 block tile, 64x64 wave tile (4x4 fragments),
//         GLOBAL_LOAD_ASYNC_TO_LDS_B128 staging, TRIPLE-buffered LDS with
//         prefetch distance 2 and partial s_wait_asynccnt (in-order async).
// Path B (small ws): fused-conversion kernel, 128x128 tile, double buffer.
// ---------------------------------------------------------------------------

typedef __attribute__((ext_vector_type(16))) __bf16 v16bf;
typedef __attribute__((ext_vector_type(8)))  float  v8f;

#define M_TOT 8192
#define N_TOT 3072
#define K_TOT 3072
#define BK 32
#define LDSS 40              // LDS row stride in halves (32 data + 8 pad)

// Path A geometry
#define BM1 128
#define BN1 256
#define ABUF1 (BM1 * LDSS)   // 5120 halves = 10 KB
#define BBUF1 (BN1 * LDSS)   // 10240 halves = 20 KB
// Path B geometry
#define BM2 128
#define BN2 128
#define ABUF2 (BM2 * LDSS)
#define BBUF2 (BN2 * LDSS)

__device__ __forceinline__ float bf2f(unsigned short h) {
    return __uint_as_float(((unsigned int)h) << 16);
}

__device__ __forceinline__ unsigned short f2bf(float f) {
    unsigned int u = __float_as_uint(f);
    u += 0x7FFFu + ((u >> 16) & 1u);      // RNE
    return (unsigned short)(u >> 16);
}

// 4 packed fp8(e4m3) -> 4 bf16 (exact, incl. fp8 denormals)
__device__ __forceinline__ uint2 cvt_fp8x4_bf16x4(unsigned int v) {
    unsigned int o[4];
#pragma unroll
    for (int i = 0; i < 4; ++i) {
        unsigned int b   = (v >> (8 * i)) & 0xFFu;
        unsigned int mag = (b & 0x7Fu) << 20;
        float f = __uint_as_float(mag) * 0x1.0p+120f;
        o[i] = (__float_as_uint(f) >> 16) | ((b & 0x80u) << 8);
    }
    uint2 r;
    r.x = o[0] | (o[1] << 16);
    r.y = o[2] | (o[3] << 16);
    return r;
}

__device__ __forceinline__ unsigned lds_addr_of(const void* p) {
    return (unsigned)(unsigned long long)
        (__attribute__((address_space(3))) const void*)p;
}

// GLOBAL_LOAD_ASYNC_TO_LDS_B128, GVS mode: mem = SADDR + VADDR(u32)
__device__ __forceinline__ void async_load_b128(unsigned lds_byte_addr,
                                                unsigned goff_bytes,
                                                const void* sbase) {
    asm volatile("global_load_async_to_lds_b128 %0, %1, %2"
                 :: "v"(lds_byte_addr), "v"(goff_bytes), "s"(sbase)
                 : "memory");
}

template <int N>
__device__ __forceinline__ void wait_async() {
#if __has_builtin(__builtin_amdgcn_s_wait_asynccnt)
    __builtin_amdgcn_s_wait_asynccnt(N);
#else
    asm volatile("s_wait_asynccnt %0" :: "i"(N) : "memory");
#endif
}

// ===========================================================================
// Kernel 0: one-shot W upcast  fp8[N,K] -> bf16[N,K]
// ===========================================================================
__global__ __launch_bounds__(256)
void w_upcast_kernel(const unsigned char* __restrict__ w,
                     unsigned short*      __restrict__ wb) {
    const size_t i = ((size_t)blockIdx.x * 256 + threadIdx.x) * 16;
    uint4 v = *(const uint4*)(w + i);
    uint2 c0 = cvt_fp8x4_bf16x4(v.x);
    uint2 c1 = cvt_fp8x4_bf16x4(v.y);
    uint2 c2 = cvt_fp8x4_bf16x4(v.z);
    uint2 c3 = cvt_fp8x4_bf16x4(v.w);
    *(uint4*)(wb + i)     = make_uint4(c0.x, c0.y, c1.x, c1.y);
    *(uint4*)(wb + i + 8) = make_uint4(c2.x, c2.y, c3.x, c3.y);
}

// ===========================================================================
// Kernel 1 (preferred): pure-bf16 GEMM, 64x64 wave tiles, triple-buffered
// async->LDS staging with prefetch distance 2.
// ===========================================================================
__global__ __launch_bounds__(256)
void bf16_gemm_wmma_kernel(const unsigned short* __restrict__ x,    // bf16 [M,K]
                           const unsigned short* __restrict__ wb,   // bf16 [N,K]
                           const unsigned short* __restrict__ bias, // bf16 [N]
                           unsigned short*       __restrict__ out)  // bf16 [M,N]
{
    __shared__ __align__(16) unsigned short lsA[3 * ABUF1];   // 30 KB
    __shared__ __align__(16) unsigned short lsB[3 * BBUF1];   // 60 KB

    const int tid  = threadIdx.x;
    const int lane = tid & 31;
    const int wid  = tid >> 5;
    const int wm   = wid & 1;           // 2 wave-rows x 64 M
    const int wn   = wid >> 1;          // 4 wave-cols x 64 N
    const int lr   = lane & 15;
    const int lh   = lane >> 4;

    const int block_n = blockIdx.x * BN1;
    const int block_m = blockIdx.y * BM1;

    // A: 128 rows x 64B = 512 x 16B chunks -> 2/thread
    // B: 256 rows x 64B = 1024 x 16B chunks -> 4/thread
    const int t_row  = tid >> 2;                 // 0..63
    const int t_koff = (tid & 3) * 8;            // halves
    unsigned gA[2], gB[4];
    int      sA[2], sB[4];
#pragma unroll
    for (int j = 0; j < 2; ++j) {
        gA[j] = (unsigned)(((size_t)(block_m + t_row + 64 * j) * K_TOT + t_koff) * 2);
        sA[j] = (t_row + 64 * j) * LDSS + t_koff;
    }
#pragma unroll
    for (int j = 0; j < 4; ++j) {
        gB[j] = (unsigned)(((size_t)(block_n + t_row + 64 * j) * K_TOT + t_koff) * 2);
        sB[j] = (t_row + 64 * j) * LDSS + t_koff;
    }

    // Fragment LDS read offsets (WMMA VGPR layouts)
    int aoff[4], boff[4];
#pragma unroll
    for (int fm = 0; fm < 4; ++fm)
        aoff[fm] = (wm * 64 + fm * 16 + lr) * LDSS + lh * 16;
#pragma unroll
    for (int fn = 0; fn < 4; ++fn)
        boff[fn] = (wn * 64 + fn * 16 + lr) * LDSS + lh * 16;

    v8f acc[4][4];
#pragma unroll
    for (int fm = 0; fm < 4; ++fm)
#pragma unroll
        for (int fn = 0; fn < 4; ++fn)
            acc[fm][fn] = {};

    // issue 6 async b128 loads (per wave) for one K-tile into a buffer
    auto issue_tile = [&](unsigned short* aDst, unsigned short* bDst, int k) {
        const unsigned kb = (unsigned)k * 2;
#pragma unroll
        for (int j = 0; j < 2; ++j)
            async_load_b128(lds_addr_of(aDst + sA[j]), gA[j] + kb, x);
#pragma unroll
        for (int j = 0; j < 4; ++j)
            async_load_b128(lds_addr_of(bDst + sB[j]), gB[j] + kb, wb);
    };

    // fragment loads + 16 WMMAs from one buffer
    auto compute = [&](const unsigned short* aCur, const unsigned short* bCur) {
        v16bf bfrag[4];
#pragma unroll
        for (int fn = 0; fn < 4; ++fn) {
            uint4* p = (uint4*)&bfrag[fn];
            const unsigned short* base = bCur + boff[fn];
            p[0] = *(const uint4*)(base);          // K = lh*16 .. +7
            p[1] = *(const uint4*)(base + 8);      // K = lh*16+8 .. +15
        }
#pragma unroll
        for (int fm = 0; fm < 4; ++fm) {
            v16bf afrag;
            uint4* p = (uint4*)&afrag;
            const unsigned short* base = aCur + aoff[fm];
            p[0] = *(const uint4*)(base);          // K = lh*16 .. +7
            p[1] = *(const uint4*)(base + 8);      // K = lh*16+8 .. +15
#pragma unroll
            for (int fn = 0; fn < 4; ++fn)
                acc[fm][fn] = __builtin_amdgcn_wmma_f32_16x16x32_bf16(
                    false, afrag, false, bfrag[fn],
                    (short)0, acc[fm][fn], false, false);
        }
    };

    unsigned short* A0 = lsA;              unsigned short* B0 = lsB;
    unsigned short* A1 = lsA + ABUF1;      unsigned short* B1 = lsB + BBUF1;
    unsigned short* A2 = lsA + 2 * ABUF1;  unsigned short* B2 = lsB + 2 * BBUF1;

    // steady-state stage: compute tile t from (cA,cB), prefetch tile t+2 into
    // (pA,pB); wait_async<6> proves tile t+1 resident (async completes in order)
    auto stageP = [&](const unsigned short* cA, const unsigned short* cB,
                      unsigned short* pA, unsigned short* pB, int kPref) {
        issue_tile(pA, pB, kPref);
        compute(cA, cB);
        wait_async<6>();
        __syncthreads();
    };

    // prologue: tiles 0 and 1 in flight; tile 0 resident after wait<6>
    issue_tile(A0, B0, 0);
    issue_tile(A1, B1, BK);
    wait_async<6>();
    __syncthreads();

    // stages 0..92 (31 branch-free triples), each prefetching distance 2
    for (int t = 0; t < 93; t += 3) {
        const int k = t * BK;
        stageP(A0, B0, A2, B2, k + 2 * BK);
        stageP(A1, B1, A0, B0, k + 3 * BK);
        stageP(A2, B2, A1, B1, k + 4 * BK);
    }
    // stage 93: prefetch last tile (95)
    stageP(A0, B0, A2, B2, 95 * BK);
    // stage 94: nothing left to prefetch; drain tile 95
    compute(A1, B1);
    wait_async<0>();
    __syncthreads();
    // stage 95
    compute(A2, B2);

    // epilogue: + bias, f32 -> bf16, store
    float bv[4];
#pragma unroll
    for (int fn = 0; fn < 4; ++fn)
        bv[fn] = bf2f(bias[block_n + wn * 64 + fn * 16 + lr]);
#pragma unroll
    for (int fm = 0; fm < 4; ++fm) {
        const int m0 = block_m + wm * 64 + fm * 16 + lh * 8;
#pragma unroll
        for (int fn = 0; fn < 4; ++fn) {
            const int n = block_n + wn * 64 + fn * 16 + lr;
            unsigned short* o = out + (size_t)m0 * N_TOT + n;
#pragma unroll
            for (int i = 0; i < 8; ++i)
                o[(size_t)i * N_TOT] = f2bf(acc[fm][fn][i] + bv[fn]);
        }
    }
}

// ===========================================================================
// Kernel 2 (fallback, ws too small): fused fp8->bf16 conversion in-loop.
// ===========================================================================
__global__ __launch_bounds__(256)
void fp8_linear_wmma_kernel(const unsigned short* __restrict__ x,
                            const unsigned char*  __restrict__ w,
                            const unsigned short* __restrict__ bias,
                            unsigned short*       __restrict__ out)
{
    __shared__ __align__(16) unsigned short lsA[2 * ABUF2];
    __shared__ __align__(16) unsigned short lsB[2 * BBUF2];

    const int tid  = threadIdx.x;
    const int lane = tid & 31;
    const int wid  = tid >> 5;
    const int wm   = wid & 3;
    const int wn   = wid >> 2;
    const int lr   = lane & 15;
    const int lh   = lane >> 4;

    const int block_n = blockIdx.x * BN2;
    const int block_m = blockIdx.y * BM2;

    const int a_row  = tid >> 2;
    const int a_koff = (tid & 3) * 8;
    const unsigned a_goff0 = (unsigned)(((size_t)(block_m + a_row)      * K_TOT + a_koff) * 2);
    const unsigned a_goff1 = (unsigned)(((size_t)(block_m + a_row + 64) * K_TOT + a_koff) * 2);
    const int sA0 = a_row * LDSS + a_koff;
    const int sA1 = sA0 + 64 * LDSS;
    const int b_row  = tid >> 1;
    const int b_koff = (tid & 1) * 16;
    const unsigned char* wg = w + (size_t)(block_n + b_row) * K_TOT + b_koff;
    const int sB0 = b_row * LDSS + b_koff;

    int aoff[2], boff[4];
#pragma unroll
    for (int fm = 0; fm < 2; ++fm)
        aoff[fm] = (wm * 32 + fm * 16 + lr) * LDSS + lh * 16;
#pragma unroll
    for (int fn = 0; fn < 4; ++fn)
        boff[fn] = (wn * 64 + fn * 16 + lr) * LDSS + lh * 16;

    v8f acc[2][4];
#pragma unroll
    for (int fm = 0; fm < 2; ++fm)
#pragma unroll
        for (int fn = 0; fn < 4; ++fn)
            acc[fm][fn] = {};

    auto stage = [&](const unsigned short* aCur, const unsigned short* bCur,
                     unsigned short* aNxt, unsigned short* bNxt,
                     int kNext, bool pf) {
        uint4 vb;
        if (pf) {
            async_load_b128(lds_addr_of(aNxt + sA0), a_goff0 + (unsigned)kNext * 2, x);
            async_load_b128(lds_addr_of(aNxt + sA1), a_goff1 + (unsigned)kNext * 2, x);
            vb = *(const uint4*)(wg + kNext);
        }

        v16bf afrag[2], bfrag[4];
#pragma unroll
        for (int fm = 0; fm < 2; ++fm) {
            uint4* p = (uint4*)&afrag[fm];
            const unsigned short* base = aCur + aoff[fm];
            p[0] = *(const uint4*)(base);
            p[1] = *(const uint4*)(base + 8);
        }
#pragma unroll
        for (int fn = 0; fn < 4; ++fn) {
            uint4* p = (uint4*)&bfrag[fn];
            const unsigned short* base = bCur + boff[fn];
            p[0] = *(const uint4*)(base);
            p[1] = *(const uint4*)(base + 8);
        }

#pragma unroll
        for (int fm = 0; fm < 2; ++fm)
#pragma unroll
            for (int fn = 0; fn < 4; ++fn)
                acc[fm][fn] = __builtin_amdgcn_wmma_f32_16x16x32_bf16(
                    false, afrag[fm], false, bfrag[fn],
                    (short)0, acc[fm][fn], false, false);

        if (pf) {
            uint2 c0 = cvt_fp8x4_bf16x4(vb.x);
            uint2 c1 = cvt_fp8x4_bf16x4(vb.y);
            uint2 c2 = cvt_fp8x4_bf16x4(vb.z);
            uint2 c3 = cvt_fp8x4_bf16x4(vb.w);
            *(uint4*)(bNxt + sB0)     = make_uint4(c0.x, c0.y, c1.x, c1.y);
            *(uint4*)(bNxt + sB0 + 8) = make_uint4(c2.x, c2.y, c3.x, c3.y);
            wait_async<0>();
        }
        __syncthreads();
    };

    {
        async_load_b128(lds_addr_of(lsA + sA0), a_goff0, x);
        async_load_b128(lds_addr_of(lsA + sA1), a_goff1, x);
        uint4 vb = *(const uint4*)(wg);
        uint2 c0 = cvt_fp8x4_bf16x4(vb.x);
        uint2 c1 = cvt_fp8x4_bf16x4(vb.y);
        uint2 c2 = cvt_fp8x4_bf16x4(vb.z);
        uint2 c3 = cvt_fp8x4_bf16x4(vb.w);
        *(uint4*)(lsB + sB0)     = make_uint4(c0.x, c0.y, c1.x, c1.y);
        *(uint4*)(lsB + sB0 + 8) = make_uint4(c2.x, c2.y, c3.x, c3.y);
        wait_async<0>();
        __syncthreads();
    }

    for (int k0 = 0; k0 < K_TOT; k0 += 2 * BK) {
        stage(lsA,         lsB,         lsA + ABUF2, lsB + BBUF2,
              k0 + BK,     (k0 + BK)     < K_TOT);
        stage(lsA + ABUF2, lsB + BBUF2, lsA,         lsB,
              k0 + 2 * BK, (k0 + 2 * BK) < K_TOT);
    }

#pragma unroll
    for (int fm = 0; fm < 2; ++fm) {
#pragma unroll
        for (int fn = 0; fn < 4; ++fn) {
            const int n  = block_n + wn * 64 + fn * 16 + lr;
            const int m0 = block_m + wm * 32 + fm * 16 + lh * 8;
            const float bv = bf2f(bias[n]);
            unsigned short* o = out + (size_t)m0 * N_TOT + n;
#pragma unroll
            for (int i = 0; i < 8; ++i)
                o[(size_t)i * N_TOT] = f2bf(acc[fm][fn][i] + bv);
        }
    }
}

extern "C" void kernel_launch(void* const* d_in, const int* in_sizes, int n_in,
                              void* d_out, int out_size, void* d_ws, size_t ws_size,
                              hipStream_t stream) {
    const unsigned short* x    = (const unsigned short*)d_in[0];
    const unsigned char*  w    = (const unsigned char*) d_in[1];
    const unsigned short* bias = (const unsigned short*)d_in[2];
    unsigned short*       out  = (unsigned short*)d_out;

    const size_t wb_bytes = (size_t)N_TOT * K_TOT * 2;   // 18.9 MB
    if (ws_size >= wb_bytes) {
        unsigned short* wb = (unsigned short*)d_ws;
        w_upcast_kernel<<<dim3((N_TOT * K_TOT) / (256 * 16)), dim3(256), 0, stream>>>(w, wb);
        bf16_gemm_wmma_kernel<<<dim3(N_TOT / BN1, M_TOT / BM1), dim3(256), 0, stream>>>(
            x, wb, bias, out);
    } else {
        fp8_linear_wmma_kernel<<<dim3(N_TOT / BN2, M_TOT / BM2), dim3(256), 0, stream>>>(
            x, w, bias, out);
    }
}
